// MAGNALayer_19542101197275
// MI455X (gfx1250) — compile-verified, matrix operand order checked by reading, not run on hardware
//
#include <hip/hip_runtime.h>

#define NN 100000
#define EE 3200000
#define NODE_BLOCKS ((NN + 63) / 64)
#define ALPHA_C 0.15f
#define ONE_M_ALPHA 0.85f

typedef __attribute__((ext_vector_type(16))) _Float16 v16h;
typedef __attribute__((ext_vector_type(8)))  float    v8f;

// K index for 16-bit A/B fragment element e (0..15) in lane-group g (0/1).
// Per ISA 7.12.2: VGPR j<4 -> K = 8g + 2j + r ; VGPR j>=4 -> K = 16 + 8g + 2(j-4) + r
__device__ __forceinline__ int wk(int e, int g) {
  int j = e >> 1, r = e & 1;
  return (j < 4) ? (8 * g + 2 * j + r) : (16 + 8 * g + 2 * (j - 4) + r);
}

// Load a 16x32 (A) or 32x16 (B, as rows of W so B[k][n] = W[n][k]) fragment.
__device__ __forceinline__ v16h frag_ld(const _Float16* p, int r0, int S, int kb, int lane) {
  int rr = r0 + (lane & 15);
  int g  = lane >> 4;
  const _Float16* q = p + (size_t)rr * S + kb;
  v16h f;
#pragma unroll
  for (int e = 0; e < 16; ++e) f[e] = q[wk(e, g)];
  return f;
}

__device__ __forceinline__ v8f wmma16(v16h a, v16h b, v8f c) {
  return __builtin_amdgcn_wmma_f32_16x16x32_f16(false, a, false, b, (short)0, c, false, false);
}

// gfx1250 async bulk copy: per-lane 16B global -> LDS, tracked by ASYNCcnt.
__device__ __forceinline__ void async_ld16(unsigned lds_off, const void* gaddr) {
  asm volatile("global_load_async_to_lds_b128 %0, %1, off"
               :: "v"(lds_off), "v"(gaddr) : "memory");
}
__device__ __forceinline__ void async_wait0() {
  asm volatile("s_wait_asynccnt 0" ::: "memory");
}

// ---------------- weight f32 -> f16 convert ----------------
__global__ __launch_bounds__(256) void k_cvt(const float* __restrict__ s,
                                             _Float16* __restrict__ d, int n) {
  int i = blockIdx.x * 256 + threadIdx.x;
  if (i < n) d[i] = (_Float16)s[i];
}

// ---------------- fused layernorm + feat = ln(x)@W_fc^T + eh/et ----------------
__global__ __launch_bounds__(256) void k_feat(const float* __restrict__ features,
                                              const _Float16* __restrict__ Wfc16,
                                              const float* __restrict__ attn_h,
                                              const float* __restrict__ attn_t,
                                              const float* __restrict__ ln_g,
                                              const float* __restrict__ ln_b,
                                              float* __restrict__ feat0,
                                              float* __restrict__ eh,
                                              float* __restrict__ et) {
  __shared__ _Float16 s_act[64][64];
  __shared__ _Float16 s_w[64][64];
  __shared__ float    s_feat[64][64];
  __shared__ float    s_sum[64][4];
  __shared__ float    s_sq[64][4];
  __shared__ float    s_ah[64];
  __shared__ float    s_at[64];

  const int tid  = threadIdx.x;
  const int base = blockIdx.x * 64;
  const int row  = tid >> 2;
  const int part = tid & 3;
  const int gn   = base + row;

  // Kick off async weight staging (8KB f16) -> overlaps with LN math below.
  {
    unsigned    wo = (unsigned)(uintptr_t)(&s_w[0][0]);
    const char* gw = (const char*)Wfc16;
    async_ld16(wo + tid * 16,         gw + tid * 16);
    async_ld16(wo + (tid + 256) * 16, gw + (tid + 256) * 16);
  }
  if (tid < 64) { s_ah[tid] = attn_h[tid]; s_at[tid] = attn_t[tid]; }

  float4 x0 = {0,0,0,0}, x1 = {0,0,0,0}, x2 = {0,0,0,0}, x3 = {0,0,0,0};
  if (gn < NN) {
    const float4* p = (const float4*)(features + (size_t)gn * 64 + part * 16);
    x0 = p[0]; x1 = p[1]; x2 = p[2]; x3 = p[3];
  }
  float s = 0.f, q = 0.f;
  const float* xv = (const float*)&x0;
#pragma unroll
  for (int i = 0; i < 16; ++i) { float v = xv[i]; s += v; q += v * v; }
  s_sum[row][part] = s; s_sq[row][part] = q;
  __syncthreads();
  float mu  = (s_sum[row][0] + s_sum[row][1] + s_sum[row][2] + s_sum[row][3]) * (1.f / 64.f);
  float var = (s_sq[row][0] + s_sq[row][1] + s_sq[row][2] + s_sq[row][3]) * (1.f / 64.f) - mu * mu;
  float rstd = rsqrtf(var + 1e-5f);
#pragma unroll
  for (int i = 0; i < 16; ++i) {
    int col = part * 16 + i;
    s_act[row][col] = (_Float16)((xv[i] - mu) * rstd * ln_g[col] + ln_b[col]);
  }
  async_wait0();       // weights landed in LDS
  __syncthreads();

  const int lane = tid & 31, wave = tid >> 5;
  for (int jb = wave; jb < 16; jb += 8) {
    int m0 = (jb >> 2) * 16, n0 = (jb & 3) * 16;
    v8f acc = {};
#pragma unroll
    for (int kb = 0; kb < 64; kb += 32) {
      v16h a = frag_ld(&s_act[0][0], m0, 64, kb, lane);
      v16h b = frag_ld(&s_w[0][0],   n0, 64, kb, lane);
      acc = wmma16(a, b, acc);
    }
    int g = lane >> 4, nn = n0 + (lane & 15);
#pragma unroll
    for (int r = 0; r < 8; ++r) s_feat[m0 + 8 * g + r][nn] = acc[r];
  }
  __syncthreads();

  // Bulk feat0 writeback + eh/et reduction (single per-thread guard).
  if (gn < NN) {
    float4*       d4 = (float4*)(feat0 + (size_t)gn * 64 + part * 16);
    const float4* s4 = (const float4*)&s_feat[row][part * 16];
    d4[0] = s4[0]; d4[1] = s4[1]; d4[2] = s4[2]; d4[3] = s4[3];
#pragma unroll
    for (int hh = 0; hh < 2; ++hh) {
      int h = part * 2 + hh;
      float se = 0.f, st = 0.f;
#pragma unroll
      for (int d = 0; d < 8; ++d) {
        float ftv = tanhf(s_feat[row][h * 8 + d]);
        se += ftv * s_ah[h * 8 + d];
        st += ftv * s_at[h * 8 + d];
      }
      eh[(size_t)gn * 8 + h] = se;
      et[(size_t)gn * 8 + h] = st;
    }
  }
}

__device__ __forceinline__ float leaky(float v) { return v >= 0.f ? v : 0.2f * v; }

__device__ __forceinline__ unsigned fkey(float v) {
  unsigned u = __float_as_uint(v);
  return ((int)u >= 0) ? (u | 0x80000000u) : ~u;
}
__device__ __forceinline__ float fdec(unsigned k) {
  return (k & 0x80000000u) ? __uint_as_float(k & 0x7FFFFFFFu) : __uint_as_float(~k);
}

// ---------------- edge softmax: segment max ----------------
__global__ __launch_bounds__(256) void k_edge_max(const int* __restrict__ src,
                                                  const int* __restrict__ dst,
                                                  const float* __restrict__ eh,
                                                  const float* __restrict__ et,
                                                  unsigned* __restrict__ emaxu) {
  int idx = blockIdx.x * 256 + threadIdx.x;
  if (idx >= EE * 8) return;
  int e = idx >> 3, h = idx & 7;
  int s = src[e], d = dst[e];
  float v = leaky(eh[(size_t)s * 8 + h] + et[(size_t)d * 8 + h]);
  atomicMax(&emaxu[(size_t)d * 8 + h], fkey(v));
}

// ---------------- edge softmax: denominator ----------------
__global__ __launch_bounds__(256) void k_edge_denom(const int* __restrict__ src,
                                                    const int* __restrict__ dst,
                                                    const float* __restrict__ eh,
                                                    const float* __restrict__ et,
                                                    const unsigned* __restrict__ emaxu,
                                                    float* __restrict__ denom) {
  int idx = blockIdx.x * 256 + threadIdx.x;
  if (idx >= EE * 8) return;
  int e = idx >> 3, h = idx & 7;
  int s = src[e], d = dst[e];
  float v  = leaky(eh[(size_t)s * 8 + h] + et[(size_t)d * 8 + h]);
  float mx = fdec(emaxu[(size_t)d * 8 + h]);
  atomicAdd(&denom[(size_t)d * 8 + h], __expf(v - mx));
}

// ---------------- PPR hop: agg[dst] += a(e,h) * f[src] (scatter, L2 resident) --
__global__ __launch_bounds__(256) void k_hop(const int* __restrict__ src,
                                             const int* __restrict__ dst,
                                             const float* __restrict__ eh,
                                             const float* __restrict__ et,
                                             const unsigned* __restrict__ emaxu,
                                             const float* __restrict__ denom,
                                             const float* __restrict__ feat0,
                                             const float* __restrict__ prev,
                                             float* __restrict__ agg,
                                             int first) {
  const int lane = threadIdx.x & 31;
  const int ew   = threadIdx.x >> 5;
  const int e    = blockIdx.x * 8 + ew;
  if (e >= EE) return;

  const int s = src[e], d = dst[e];
  const int h = lane >> 2;            // 2 feats/lane share a head
  float sc = leaky(eh[(size_t)s * 8 + h] + et[(size_t)d * 8 + h]);
  float mx = fdec(emaxu[(size_t)d * 8 + h]);
  float a  = __expf(sc - mx) / denom[(size_t)d * 8 + h];

  // prefetch a future source row (global_prefetch path)
  int e2 = e + 512;
  if (lane == 0 && e2 < EE) {
    const float* fb = first ? feat0 : prev;
    __builtin_prefetch(fb + (size_t)src[e2] * 64, 0, 1);
  }

  const int c0 = lane * 2;
  float f0, f1;
  if (first) {
    float2 fv = *(const float2*)(feat0 + (size_t)s * 64 + c0);
    f0 = fv.x; f1 = fv.y;
  } else {
    float2 pv = *(const float2*)(prev  + (size_t)s * 64 + c0);
    float2 zv = *(const float2*)(feat0 + (size_t)s * 64 + c0);
    f0 = fmaf(ONE_M_ALPHA, pv.x, ALPHA_C * zv.x);
    f1 = fmaf(ONE_M_ALPHA, pv.y, ALPHA_C * zv.y);
  }
  atomicAdd(&agg[(size_t)d * 64 + c0],     a * f0);
  atomicAdd(&agg[(size_t)d * 64 + c0 + 1], a * f1);
}

// ---------------- rst = f @ W_out^T + features  (f folded from agg/feat0) -----
__global__ __launch_bounds__(256) void k_rst(const float* __restrict__ features,
                                             const float* __restrict__ feat0,
                                             const float* __restrict__ flast,
                                             const _Float16* __restrict__ Wout16,
                                             float* __restrict__ out) {
  __shared__ _Float16 s_act[64][64];
  __shared__ _Float16 s_w[64][64];
  __shared__ float    s_feat[64][64];

  const int tid  = threadIdx.x;
  const int base = blockIdx.x * 64;
  const int row  = tid >> 2, part = tid & 3;
  const int gn   = base + row;

  {
    unsigned    wo = (unsigned)(uintptr_t)(&s_w[0][0]);
    const char* gw = (const char*)Wout16;
    async_ld16(wo + tid * 16,         gw + tid * 16);
    async_ld16(wo + (tid + 256) * 16, gw + (tid + 256) * 16);
  }

  if (gn < NN) {
    const float4* pA = (const float4*)(flast + (size_t)gn * 64 + part * 16);
    const float4* pZ = (const float4*)(feat0 + (size_t)gn * 64 + part * 16);
#pragma unroll
    for (int k = 0; k < 4; ++k) {
      float4 av = pA[k], zv = pZ[k];
      int c = part * 16 + k * 4;
      s_act[row][c + 0] = (_Float16)fmaf(ONE_M_ALPHA, av.x, ALPHA_C * zv.x);
      s_act[row][c + 1] = (_Float16)fmaf(ONE_M_ALPHA, av.y, ALPHA_C * zv.y);
      s_act[row][c + 2] = (_Float16)fmaf(ONE_M_ALPHA, av.z, ALPHA_C * zv.z);
      s_act[row][c + 3] = (_Float16)fmaf(ONE_M_ALPHA, av.w, ALPHA_C * zv.w);
    }
  } else {
#pragma unroll
    for (int i = 0; i < 16; ++i) s_act[row][part * 16 + i] = (_Float16)0.f;
  }
  async_wait0();
  __syncthreads();

  const int lane = tid & 31, wave = tid >> 5;
  for (int jb = wave; jb < 16; jb += 8) {
    int m0 = (jb >> 2) * 16, n0 = (jb & 3) * 16;
    v8f acc = {};
#pragma unroll
    for (int kb = 0; kb < 64; kb += 32) {
      v16h a = frag_ld(&s_act[0][0], m0, 64, kb, lane);
      v16h b = frag_ld(&s_w[0][0],   n0, 64, kb, lane);
      acc = wmma16(a, b, acc);
    }
    int g = lane >> 4, nn = n0 + (lane & 15);
#pragma unroll
    for (int r = 0; r < 8; ++r) s_feat[m0 + 8 * g + r][nn] = acc[r];
  }
  __syncthreads();

  if (gn < NN) {
    const float4* pF = (const float4*)(features + (size_t)gn * 64 + part * 16);
    const float4* s4 = (const float4*)&s_feat[row][part * 16];
    float4*       d4 = (float4*)(out + (size_t)gn * 64 + part * 16);
#pragma unroll
    for (int k = 0; k < 4; ++k) {
      float4 a = s4[k], b = pF[k];
      float4 r; r.x = a.x + b.x; r.y = a.y + b.y; r.z = a.z + b.z; r.w = a.w + b.w;
      d4[k] = r;   // res_fc = Identity
    }
  }
}

// ---------------- FFN: out += relu(ln(out)@w1^T+b1)@w2^T + b2 ----------------
__global__ __launch_bounds__(256) void k_ff(float* __restrict__ out,
                                            const float* __restrict__ ff_g,
                                            const float* __restrict__ ff_b,
                                            const _Float16* __restrict__ w116,
                                            const float* __restrict__ b1,
                                            const _Float16* __restrict__ w216,
                                            const float* __restrict__ b2) {
  __shared__ _Float16 s_act[64][64];
  __shared__ _Float16 s_t1[64][256];
  __shared__ float    s_sum[64][4];
  __shared__ float    s_sq[64][4];

  const int tid  = threadIdx.x;
  const int base = blockIdx.x * 64;
  const int row  = tid >> 2, part = tid & 3;
  const int gn   = base + row;
  const bool full = (base + 64 <= NN);

  float4 x0 = {0,0,0,0}, x1 = {0,0,0,0}, x2 = {0,0,0,0}, x3 = {0,0,0,0};
  if (gn < NN) {
    const float4* p = (const float4*)(out + (size_t)gn * 64 + part * 16);
    x0 = p[0]; x1 = p[1]; x2 = p[2]; x3 = p[3];
  }
  float s = 0.f, q = 0.f;
  const float* xv = (const float*)&x0;
#pragma unroll
  for (int i = 0; i < 16; ++i) { float v = xv[i]; s += v; q += v * v; }
  s_sum[row][part] = s; s_sq[row][part] = q;
  __syncthreads();
  float mu  = (s_sum[row][0] + s_sum[row][1] + s_sum[row][2] + s_sum[row][3]) * (1.f / 64.f);
  float var = (s_sq[row][0] + s_sq[row][1] + s_sq[row][2] + s_sq[row][3]) * (1.f / 64.f) - mu * mu;
  float rstd = rsqrtf(var + 1e-5f);
#pragma unroll
  for (int i = 0; i < 16; ++i) {
    int col = part * 16 + i;
    s_act[row][col] = (_Float16)((xv[i] - mu) * rstd * ff_g[col] + ff_b[col]);
  }
  __syncthreads();

  const int lane = tid & 31, wave = tid >> 5;
  // GEMM1: [64x64] x [64x256] -> t1, relu(+b1), f16 in LDS
  for (int jb = wave; jb < 64; jb += 8) {
    int m0 = (jb >> 4) * 16, n0 = (jb & 15) * 16;
    v8f acc = {};
#pragma unroll
    for (int kb = 0; kb < 64; kb += 32) {
      v16h a = frag_ld(&s_act[0][0], m0, 64, kb, lane);
      v16h b = frag_ld(w116,         n0, 64, kb, lane);
      acc = wmma16(a, b, acc);
    }
    int g = lane >> 4, nn = n0 + (lane & 15);
    float bb = b1[nn];
#pragma unroll
    for (int r = 0; r < 8; ++r) {
      float v = acc[r] + bb;
      s_t1[m0 + 8 * g + r][nn] = (_Float16)(v > 0.f ? v : 0.f);
    }
  }
  __syncthreads();

  // GEMM2: [64x256] x [256x64] -> ff; out += ff + b2 (in place, disjoint elems)
  for (int jb = wave; jb < 16; jb += 8) {
    int m0 = (jb >> 2) * 16, n0 = (jb & 3) * 16;
    v8f acc = {};
#pragma unroll
    for (int kb = 0; kb < 256; kb += 32) {
      v16h a = frag_ld(&s_t1[0][0], m0, 256, kb, lane);
      v16h b = frag_ld(w216,        n0, 256, kb, lane);
      acc = wmma16(a, b, acc);
    }
    int g = lane >> 4, nn = n0 + (lane & 15);
    float bb = b2[nn];
    if (full) {
#pragma unroll
      for (int r = 0; r < 8; ++r) {
        size_t idx = (size_t)(base + m0 + 8 * g + r) * 64 + nn;
        out[idx] = out[idx] + acc[r] + bb;
      }
    } else {
#pragma unroll
      for (int r = 0; r < 8; ++r) {
        int gm = base + m0 + 8 * g + r;
        if (gm < NN) {
          size_t idx = (size_t)gm * 64 + nn;
          out[idx] = out[idx] + acc[r] + bb;
        }
      }
    }
  }
}

extern "C" void kernel_launch(void* const* d_in, const int* in_sizes, int n_in,
                              void* d_out, int out_size, void* d_ws, size_t ws_size,
                              hipStream_t stream) {
  (void)in_sizes; (void)n_in; (void)out_size; (void)ws_size;
  const float* features = (const float*)d_in[0];
  const int*   src      = (const int*)d_in[1];
  const int*   dst      = (const int*)d_in[2];
  const float* W_fc     = (const float*)d_in[3];
  const float* attn_h   = (const float*)d_in[4];
  const float* attn_t   = (const float*)d_in[5];
  const float* W_out    = (const float*)d_in[6];
  const float* ln_g     = (const float*)d_in[7];
  const float* ln_b     = (const float*)d_in[8];
  const float* ff_g     = (const float*)d_in[9];
  const float* ff_b     = (const float*)d_in[10];
  const float* w1       = (const float*)d_in[11];
  const float* b1       = (const float*)d_in[12];
  const float* w2       = (const float*)d_in[13];
  const float* b2       = (const float*)d_in[14];
  float* out = (float*)d_out;

  const size_t N64 = (size_t)NN * 64, N8 = (size_t)NN * 8;
  float*    feat0  = (float*)d_ws;
  float*    bufA   = feat0 + N64;
  float*    bufB   = bufA + N64;
  float*    eh     = bufB + N64;
  float*    et     = eh + N8;
  unsigned* emaxu  = (unsigned*)(et + N8);
  float*    denom  = (float*)(emaxu + N8);
  _Float16* Wfc16  = (_Float16*)(denom + N8);
  _Float16* Wout16 = Wfc16 + 4096;
  _Float16* w116   = Wout16 + 4096;
  _Float16* w216   = w116 + 16384;

  k_cvt<<<16, 256, 0, stream>>>(W_fc, Wfc16, 4096);
  k_cvt<<<16, 256, 0, stream>>>(W_out, Wout16, 4096);
  k_cvt<<<64, 256, 0, stream>>>(w1, w116, 16384);
  k_cvt<<<64, 256, 0, stream>>>(w2, w216, 16384);

  hipMemsetAsync(emaxu, 0, N8 * 4, stream);
  hipMemsetAsync(denom, 0, N8 * 4, stream);

  k_feat<<<NODE_BLOCKS, 256, 0, stream>>>(features, Wfc16, attn_h, attn_t,
                                          ln_g, ln_b, feat0, eh, et);

  const int eg = (EE * 8 + 255) / 256;
  k_edge_max<<<eg, 256, 0, stream>>>(src, dst, eh, et, emaxu);
  k_edge_denom<<<eg, 256, 0, stream>>>(src, dst, eh, et, emaxu, denom);

  float* prev = nullptr;
  float* cur  = bufA;
  for (int hop = 0; hop < 5; ++hop) {
    hipMemsetAsync(cur, 0, N64 * 4, stream);
    k_hop<<<(EE + 7) / 8, 256, 0, stream>>>(src, dst, eh, et, emaxu, denom,
                                            feat0, prev, cur, hop == 0 ? 1 : 0);
    prev = cur;
    cur  = (cur == bufA) ? bufB : bufA;
  }

  k_rst<<<NODE_BLOCKS, 256, 0, stream>>>(features, feat0, prev, Wout16, out);
  k_ff<<<NODE_BLOCKS, 256, 0, stream>>>(out, ff_g, ff_b, w116, b1, w216, b2);
}
